// HPNN_83219286327449
// MI455X (gfx1250) — matrix operand order repaired and checked by reading, not compiled, not run on hardware
//
#include <hip/hip_runtime.h>
#include <hip/hip_bf16.h>

// CDNA5 / gfx1250 wave32 WMMA types
typedef __attribute__((ext_vector_type(16))) __bf16 bf16x16;
typedef __attribute__((ext_vector_type(8)))  __bf16 bf16x8;
typedef __attribute__((ext_vector_type(8)))  float  v8f;

#define HS_LAMBDA 1e-6f
#define E_DIM 128
#define N_NODES 7000
#define N_DRUG 6000
#define N_SE 1000
#define M_TRIPLES 300000

__device__ inline __bf16 to_bf16(float f) { return (__bf16)f; }

__device__ inline v8f zero_v8f() {
    v8f z;
#pragma unroll
    for (int i = 0; i < 8; ++i) z[i] = 0.0f;
    return z;
}

__device__ inline float hardshrink_f(float v) {
    return (fabsf(v) > HS_LAMBDA) ? v : 0.0f;
}

// ---- CDNA5 async global->LDS copy (ASYNCcnt-tracked, bypasses VGPRs) ------
// lds_addr: 32-bit LDS byte address (low 32 bits of the generic pointer),
// src: 64-bit global address. GV addressing mode.
__device__ inline void async_tile_b128(unsigned lds_addr, const float* src) {
    asm volatile("global_load_async_to_lds_b128 %0, %1, off"
                 :: "v"(lds_addr), "v"(src) : "memory");
}
__device__ inline void wait_async_le1() {
    asm volatile("s_wait_asynccnt 0x1" ::: "memory");
}
__device__ inline void wait_async_le0() {
    asm volatile("s_wait_asynccnt 0x0" ::: "memory");
}

// B fragment loader. Layout (wave32 16x16x32 bf16):
//   lane&15 = N column, elem e -> K = kbase + 8*(lane>>4) + e (e<8), kbase + 16 + 8*(lane>>4) + (e-8) (e>=8).
template <bool BF16B>
__device__ inline bf16x16 load_b_frag(const void* W, int kbase, int col0, int klim) {
    bf16x16 b;
    const int l  = threadIdx.x & 31;
    const int n  = col0 + (l & 15);
    const int kh = (l >> 4) * 8;
#pragma unroll
    for (int e = 0; e < 16; ++e) {
        int k = kbase + ((e < 8) ? (kh + e) : (16 + kh + e - 8));
        if (k < klim) {
            if (BF16B) b[e] = ((const __bf16*)W)[(size_t)k * E_DIM + n];
            else       b[e] = to_bf16(((const float*)W)[(size_t)k * E_DIM + n]);
        } else {
            b[e] = (__bf16)0.0f;
        }
    }
    return b;
}

// ---------------------------------------------------------------------------
// Y[M,128] = act( X[M,K] @ W[K,128] + bias ),  optional bf16 shadow copy Ybf.
// Block = 256 threads = 8 waves; wave w owns output columns [16w, 16w+16).
// A tile [16 x 64] f32 is DMA'd into a double-buffered LDS tile with
// global_load_async_to_lds_b128 (one async op per thread per tile) and
// software-pipelined one tile ahead; s_wait_asynccnt + barrier publish it.
// All 8 waves share the tile -> A HBM traffic stays at the streaming minimum
// (196 MB per propagation layer). f32->bf16 conversion happens at fragment
// read (v_cvt_pk co-executes with WMMA). K-tail (K%64) is peeled onto a
// synchronous guarded path so async copies never read out of bounds.
// act: 0 = none, 1 = relu, 2 = hardshrink
// ---------------------------------------------------------------------------
template <bool BF16B>
__global__ __launch_bounds__(256)
void gemm128_bias_act(const float* __restrict__ X,
                      const void* __restrict__ W,
                      const float* __restrict__ bias,
                      float* __restrict__ Y,
                      __bf16* __restrict__ Ybf,
                      int M, int K, int ldx, int act) {
    __shared__ __align__(16) float As[2][16][68];  // 64 + 4 pad (row = 272 B)

    const int tid  = threadIdx.x;
    const int lane = tid & 31;
    const int wv   = tid >> 5;                    // N-block 0..7
    const int row0 = blockIdx.x * 16;

    // --- hoisted fill slot: row fm, K offset fk..fk+3 (16 B per thread) ---
    const int fm = tid >> 4;                      // 0..15
    const int fk = (tid & 15) * 4;                // 0..60
    int frow = row0 + fm; if (frow >= M) frow = M - 1;   // clamp; rows unused
    const float* __restrict__ xr = X + (size_t)frow * ldx;
    const float* gsrc = xr + fk;

    // LDS byte addresses of this thread's slot in each buffer
    const unsigned slot0 = (unsigned)(size_t)&As[0][fm][fk];
    const unsigned slot1 = (unsigned)(size_t)&As[1][fm][fk];

    // --- hoisted fragment-read coordinates ---
    const int am = lane & 15;
    const int kh = (lane >> 4) * 8;

    v8f acc = zero_v8f();
    const int Kmain = K & ~63;

    if (Kmain > 0) async_tile_b128(slot0, gsrc);  // prefetch tile 0

    int buf = 0;
    for (int k0 = 0; k0 < Kmain; k0 += 64) {
        const bool more = (k0 + 64) < Kmain;
        if (more) {
            async_tile_b128(buf ? slot0 : slot1, gsrc + (k0 + 64));
            wait_async_le1();                     // current tile landed
        } else {
            wait_async_le0();
        }
        __syncthreads();                          // all waves' rows resident
#pragma unroll
        for (int sub = 0; sub < 64; sub += 32) {
            bf16x16 a;                            // 4x ds_load_b128 + cvt_pk
#pragma unroll
            for (int e = 0; e < 8; ++e) a[e]     = to_bf16(As[buf][am][sub + kh + e]);
#pragma unroll
            for (int e = 0; e < 8; ++e) a[e + 8] = to_bf16(As[buf][am][sub + 16 + kh + e]);
            bf16x16 bf = load_b_frag<BF16B>(W, k0 + sub, wv * 16, K);
            acc = __builtin_amdgcn_wmma_f32_16x16x32_bf16(
                      false, a, false, bf, (short)0, acc, false, false);
        }
        __syncthreads();                          // reads done before reuse
        buf ^= 1;
    }

    if (Kmain < K) {                              // peeled guarded tail (sync)
        const int tail = K - Kmain;
        float tv[4];
#pragma unroll
        for (int e = 0; e < 4; ++e) {
            int kk = Kmain + fk + e;
            float v = xr[(kk < K) ? kk : 0];
            tv[e] = (kk < K) ? v : 0.0f;
        }
        __syncthreads();
#pragma unroll
        for (int e = 0; e < 4; ++e) As[0][fm][fk + e] = tv[e];
        __syncthreads();
        for (int sub = 0; sub < tail; sub += 32) {
            bf16x16 a;
#pragma unroll
            for (int e = 0; e < 8; ++e) a[e]     = to_bf16(As[0][am][sub + kh + e]);
#pragma unroll
            for (int e = 0; e < 8; ++e) a[e + 8] = to_bf16(As[0][am][sub + 16 + kh + e]);
            bf16x16 bf = load_b_frag<BF16B>(W, Kmain + sub, wv * 16, K);
            acc = __builtin_amdgcn_wmma_f32_16x16x32_bf16(
                      false, a, false, bf, (short)0, acc, false, false);
        }
    }

    // epilogue: C/D layout -> lane&15 = N, vgpr r -> M = r + 8*(lane>>4)
    const int col   = wv * 16 + (lane & 15);
    const int rbase = (lane >> 4) * 8;
    const float bv  = bias ? bias[col] : 0.0f;
#pragma unroll
    for (int r = 0; r < 8; ++r) {
        int row = row0 + rbase + r;
        if (row < M) {
            float v = acc[r] + bv;
            if (act == 1)      v = fmaxf(v, 0.0f);
            else if (act == 2) v = hardshrink_f(v);
            Y[(size_t)row * E_DIM + col] = v;
            if (Ybf) Ybf[(size_t)row * E_DIM + col] = to_bf16(v);
        }
    }
}

// ---------------------------------------------------------------------------
// Fused head: 16 triples per block. Gather + scale three x rows per triple
// into LDS as bf16 (h tile [16,384]); 12 guard-free WMMA K-steps against
// pre-converted bf16 Wp1; hardshrink + Wp2 dot via shfl_xor in-wave reduce
// and cross-wave LDS reduce; final hardshrink -> out. x (3.6 MB) and Wp1bf
// (96 KB) are L2-resident, so gathers never touch HBM.
// ---------------------------------------------------------------------------
__global__ __launch_bounds__(256)
void head_kernel(const float* __restrict__ x,
                 const long long* __restrict__ tpl,
                 const float* __restrict__ rsqrtdeg,
                 const __bf16* __restrict__ Wp1bf,
                 const float* __restrict__ bp1,
                 const float* __restrict__ Wp2,
                 const float* __restrict__ bp2,
                 float* __restrict__ out, int Mtr) {
    __shared__ __align__(16) __bf16 Hs[16][392];  // 384 + 8 pad (row = 784 B)
    __shared__ float partial[8][16];

    const int t0   = blockIdx.x * 16;
    const int tid  = threadIdx.x;
    const int lane = tid & 31;
    const int wv   = tid >> 5;

    // gather + scale: 96 quads per row, 16 rows = 1536 quads; 6 per thread
    for (int q = tid; q < 16 * 96; q += 256) {
        int m   = q / 96;
        int c4  = (q - m * 96) * 4;               // 0..380, multiple of 4
        int seg = c4 >> 7;                        // 0: d1, 1: d2, 2: se
        int col = c4 & 127;
        int trip = t0 + m; if (trip >= Mtr) trip = Mtr - 1;
        long long idx = tpl[(long long)trip * 3 + seg];
        float sc = rsqrtdeg[idx];
        const float* xr = x + (long long)idx * E_DIM + col;  // 16B load
        union { __bf16 b[4]; unsigned long long u; } pk;
#pragma unroll
        for (int e = 0; e < 4; ++e) pk.b[e] = to_bf16(xr[e] * sc);
        *(unsigned long long*)&Hs[m][c4] = pk.u;  // ds_store_b64
    }
    __syncthreads();

    const int am = lane & 15;
    const int kh = (lane >> 4) * 8;
    v8f acc = zero_v8f();
#pragma unroll
    for (int k0 = 0; k0 < 384; k0 += 32) {
        union { bf16x16 v; bf16x8 h[2]; } au;     // two ds_load_b128
        au.h[0] = *(const bf16x8*)&Hs[am][k0 + kh];
        au.h[1] = *(const bf16x8*)&Hs[am][k0 + 16 + kh];
        bf16x16 bf = load_b_frag<true>(Wp1bf, k0, wv * 16, 384);
        acc = __builtin_amdgcn_wmma_f32_16x16x32_bf16(
                  false, au.v, false, bf, (short)0, acc, false, false);
    }

    // hardshrink(hidden) . Wp2 — each lane covers 8 rows of one column
    const int col = wv * 16 + (lane & 15);
    const float b1v = bp1[col];
    const float w2v = Wp2[col];
    float p[8];
#pragma unroll
    for (int r = 0; r < 8; ++r)
        p[r] = hardshrink_f(acc[r] + b1v) * w2v;

    // reduce the 16 lanes holding the same row group (xor stays in-group)
#pragma unroll
    for (int off = 8; off >= 1; off >>= 1) {
#pragma unroll
        for (int r = 0; r < 8; ++r) p[r] += __shfl_xor(p[r], off, 32);
    }
    if ((lane & 15) == 0) {
        int rbase = (lane >> 4) * 8;
#pragma unroll
        for (int r = 0; r < 8; ++r) partial[wv][rbase + r] = p[r];
    }
    __syncthreads();

    if (tid < 16) {
        float s = 0.0f;
#pragma unroll
        for (int ww = 0; ww < 8; ++ww) s += partial[ww][tid];
        s = hardshrink_f(s + bp2[0]);
        if (t0 + tid < Mtr) out[t0 + tid] = s;
    }
}

__global__ void copy_dual(const float* __restrict__ src,
                          float* __restrict__ dst, __bf16* __restrict__ dstb, int n) {
    int i = blockIdx.x * blockDim.x + threadIdx.x;
    if (i < n) { float v = src[i]; dst[i] = v; dstb[i] = to_bf16(v); }
}

__global__ void convert_bf16_kernel(const float* __restrict__ src,
                                    __bf16* __restrict__ dst, int n) {
    int i = blockIdx.x * blockDim.x + threadIdx.x;
    if (i < n) dst[i] = to_bf16(src[i]);
}

extern "C" void kernel_launch(void* const* d_in, const int* in_sizes, int n_in,
                              void* d_out, int out_size, void* d_ws, size_t ws_size,
                              hipStream_t stream) {
    (void)in_sizes; (void)n_in; (void)out_size; (void)ws_size;

    const float*     drugFeatures = (const float*)d_in[0];
    const float*     A            = (const float*)d_in[1];
    const long long* tpl          = (const long long*)d_in[2];   // int64 in reference
    const float*     rsqrtdeg     = (const float*)d_in[3];
    const float*     W1           = (const float*)d_in[4];
    const float*     b1           = (const float*)d_in[5];
    const float*     W2           = (const float*)d_in[6];
    const float*     b2           = (const float*)d_in[7];
    const float*     embSe        = (const float*)d_in[8];
    const float*     Wl           = (const float*)d_in[9];
    const float*     bl           = (const float*)d_in[10];
    const float*     Wp1          = (const float*)d_in[11];
    const float*     bp1          = (const float*)d_in[12];
    const float*     Wp2          = (const float*)d_in[13];
    const float*     bp2          = (const float*)d_in[14];
    float*           out          = (float*)d_out;

    // workspace layout (~9.1 MB)
    float*  x     = (float*)d_ws;                          // [7000,128] f32
    float*  T     = x + (size_t)N_NODES * E_DIM;           // [7000,128] f32
    __bf16* xb    = (__bf16*)(T + (size_t)N_NODES * E_DIM);// [7000,128] bf16
    __bf16* Wp1bf = xb + (size_t)N_NODES * E_DIM;          // [384,128] bf16

    dim3 blk(256);

    // 1) T[0:6000] = relu(drugFeatures @ W1 + b1)      (K=1024)
    gemm128_bias_act<false><<<N_DRUG / 16, blk, 0, stream>>>(
        drugFeatures, W1, b1, T, nullptr, N_DRUG, 1024, 1024, /*relu*/1);

    // 2) x[0:6000] = relu(T @ W2 + b2), shadow bf16 copy into xb
    gemm128_bias_act<false><<<N_DRUG / 16, blk, 0, stream>>>(
        T, W2, b2, x, xb, N_DRUG, 128, 128, /*relu*/1);

    // 3) x[6000:7000] = embSe (f32 + bf16 shadow)
    copy_dual<<<(N_SE * E_DIM + 255) / 256, blk, 0, stream>>>(
        embSe, x + (size_t)N_DRUG * E_DIM, xb + (size_t)N_DRUG * E_DIM,
        N_SE * E_DIM);

    // 4) Wp1 -> bf16 once (shared by all 18750 head blocks via L2)
    convert_bf16_kernel<<<(384 * E_DIM + 255) / 256, blk, 0, stream>>>(
        Wp1, Wp1bf, 384 * E_DIM);

    // 5) graph propagation layers
    for (int l = 0; l < 2; ++l) {
        // T = A @ x : 196 MB A stream via async double-buffered LDS tiles
        gemm128_bias_act<true><<<(N_NODES + 15) / 16, blk, 0, stream>>>(
            A, xb, nullptr, T, nullptr, N_NODES, N_NODES, N_NODES, /*none*/0);
        // x = hardshrink(T @ Wl[l] + bl[l]), refresh bf16 shadow xb
        gemm128_bias_act<false><<<(N_NODES + 15) / 16, blk, 0, stream>>>(
            T, Wl + (size_t)l * E_DIM * E_DIM, bl + (size_t)l * E_DIM,
            x, xb, N_NODES, 128, 128, /*hardshrink*/2);
    }

    // 6) fused gather + 2-layer MLP head
    head_kernel<<<(M_TRIPLES + 15) / 16, blk, 0, stream>>>(
        x, tpl, rsqrtdeg, Wp1bf, bp1, Wp2, bp2, out, M_TRIPLES);
}
